// ProtoLoss_73873437491538
// MI455X (gfx1250) — matrix operand order repaired and checked by reading, not compiled, not run on hardware
//
#include <hip/hip_runtime.h>

typedef __attribute__((ext_vector_type(16))) _Float16 v16h;
typedef __attribute__((ext_vector_type(8)))  _Float16 v8h;
typedef __attribute__((ext_vector_type(8)))  float    v8f;

#define N_ROWS 8192
#define DFEAT  192
#define EPSF   1e-6f

// ---------------------------------------------------------------------------
// Kernel 1: preprocess. pos2 = 2*x[:,0,:], anchor = 0.5*(x[:,1,:]+x[:,2,:]).
// Split each into f16 hi + f16 lo (v = hi + lo, near-f32 accuracy), and
// compute per-column bias b[j] = -||anchor_j||^2 + 2*EPS*sum(anchor_j).
// One wave per row (8 rows per 256-thread block).
// ---------------------------------------------------------------------------
__global__ __launch_bounds__(256) void proto_prep(
    const float* __restrict__ x,
    _Float16* __restrict__ ph, _Float16* __restrict__ pl,
    _Float16* __restrict__ ah, _Float16* __restrict__ al,
    float* __restrict__ bvec)
{
  const int lane = threadIdx.x & 31;
  const int wave = threadIdx.x >> 5;
  const int row  = blockIdx.x * 8 + wave;
  const float* xr = x + (size_t)row * 3 * DFEAT;
  const size_t ro = (size_t)row * DFEAT;

  float sa = 0.f, sq = 0.f;
  #pragma unroll
  for (int d = lane; d < DFEAT; d += 32) {
    float p = 2.0f * xr[d];
    _Float16 phh = (_Float16)p;
    ph[ro + d] = phh;
    pl[ro + d] = (_Float16)(p - (float)phh);
    float a = 0.5f * (xr[DFEAT + d] + xr[2 * DFEAT + d]);
    _Float16 ahh = (_Float16)a;
    ah[ro + d] = ahh;
    al[ro + d] = (_Float16)(a - (float)ahh);
    sa += a;
    sq += a * a;
  }
  #pragma unroll
  for (int mask = 16; mask > 0; mask >>= 1) {
    sa += __shfl_xor(sa, mask, 32);
    sq += __shfl_xor(sq, mask, 32);
  }
  if (lane == 0) bvec[row] = -sq + 2.0f * EPSF * sa;
}

// ---------------------------------------------------------------------------
// Kernel 2: fused GEMM + online log-softmax stats. One wave owns a 16-row
// tile; A fragments (hi+lo, K=192) held in VGPRs for the whole j-loop; B
// tiles streamed from L2. logit[i,j] = (p2_i . anchor_j) + b[j].
// Split product: hi*hi + hi*lo + lo*hi accumulated in f32 via WMMA.
// ---------------------------------------------------------------------------
__global__ __launch_bounds__(128) void proto_gemm(
    const _Float16* __restrict__ ph, const _Float16* __restrict__ pl,
    const _Float16* __restrict__ ah, const _Float16* __restrict__ al,
    const float* __restrict__ bvec,
    float* __restrict__ lossterm, float* __restrict__ correct)
{
  const int lane = threadIdx.x & 31;
  const int wave = threadIdx.x >> 5;
  const int it   = blockIdx.x * 4 + wave;          // row-tile index 0..511
  const int mrow = lane & 15;
  const int koff = (lane & 16) ? 16 : 0;           // A-frag lane K offset
  const int boff = (lane & 16) ? 16 : 0;           // B-frag lane K offset

  // A fragments: 16-bit A 16x32 layout — lane m holds row m;
  // VGPR0-7 = K koff..koff+15 contiguous (matches B).
  v16h Ah[6], Al[6];
  {
    const _Float16* prh = ph + (size_t)(it * 16 + mrow) * DFEAT;
    const _Float16* prl = pl + (size_t)(it * 16 + mrow) * DFEAT;
    #pragma unroll
    for (int kc = 0; kc < 6; ++kc) {
      const int kb = kc * 32 + koff;
      union { v16h v; v8h h[2]; } fh, fl;
      fh.h[0] = *(const v8h*)(prh + kb);
      fh.h[1] = *(const v8h*)(prh + kb + 8);
      fl.h[0] = *(const v8h*)(prl + kb);
      fl.h[1] = *(const v8h*)(prl + kb + 8);
      Ah[kc] = fh.v;
      Al[kc] = fl.v;
    }
  }

  float mrun[8], srun[8];
  int   bj[8];
  #pragma unroll
  for (int e = 0; e < 8; ++e) { mrun[e] = -3.0e38f; srun[e] = 0.f; bj[e] = 0; }
  float diagv = 0.f;

  for (int jt = 0; jt < N_ROWS / 16; ++jt) {
    // B fragment: 16-bit B 32x16 — lane holds column n=lane&15,
    // K = boff..boff+15 packed contiguously (anchor row-major => contiguous).
    const size_t brow = (size_t)(jt * 16 + mrow) * DFEAT + boff;
    const _Float16* arh = ah + brow;
    const _Float16* arl = al + brow;

    v8f acc = {};
    #pragma unroll
    for (int kc = 0; kc < 6; ++kc) {
      v16h bh = *(const v16h*)(arh + kc * 32);
      v16h bl = *(const v16h*)(arl + kc * 32);
      acc = __builtin_amdgcn_wmma_f32_16x16x32_f16(false, Ah[kc], false, bh, (short)0, acc, false, false);
      acc = __builtin_amdgcn_wmma_f32_16x16x32_f16(false, Ah[kc], false, bl, (short)0, acc, false, false);
      acc = __builtin_amdgcn_wmma_f32_16x16x32_f16(false, Al[kc], false, bh, (short)0, acc, false, false);
    }

    const int col = jt * 16 + (lane & 15);
    const float bjv = bvec[col];

    #pragma unroll
    for (int e = 0; e < 8; ++e) {
      float v = acc[e] + bjv;
      if (v > mrun[e]) {                         // new running max (rare)
        srun[e] = srun[e] * __expf(mrun[e] - v) + 1.0f;
        mrun[e] = v;
        bj[e]   = col;
      } else {                                   // common path: single exp
        srun[e] += __expf(v - mrun[e]);
      }
    }

    if (jt == it) {                              // capture diagonal logit
      if (lane < 8)        diagv = acc[lane]      + bjv;
      else if (lane >= 24) diagv = acc[lane - 24] + bjv;
    }
  }

  // Merge (max, sumexp, argmax) across the 16 lanes of each half-wave.
  #pragma unroll
  for (int mask = 1; mask < 16; mask <<= 1) {
    #pragma unroll
    for (int e = 0; e < 8; ++e) {
      float om = __shfl_xor(mrun[e], mask, 32);
      float os = __shfl_xor(srun[e], mask, 32);
      int   oj = __shfl_xor(bj[e],   mask, 32);
      float nm = fmaxf(mrun[e], om);
      srun[e] = srun[e] * __expf(mrun[e] - nm) + os * __expf(om - nm);
      if (om > mrun[e] || (om == mrun[e] && oj < bj[e])) bj[e] = oj;
      mrun[e] = nm;
    }
  }

  // Lanes 0-7 own rows 0-7 (and their diagonals); lanes 24-31 own rows 8-15.
  if (lane < 8) {
    const int e = lane, gi = it * 16 + e;
    lossterm[gi] = (mrun[e] + __logf(srun[e])) - diagv;
    correct[gi]  = (bj[e] == gi) ? 1.0f : 0.0f;
  } else if (lane >= 24) {
    const int e = lane - 24, gi = it * 16 + 8 + e;
    lossterm[gi] = (mrun[e] + __logf(srun[e])) - diagv;
    correct[gi]  = (bj[e] == gi) ? 1.0f : 0.0f;
  }
}

// ---------------------------------------------------------------------------
// Kernel 3: deterministic final reduction -> (nloss, prec1).
// ---------------------------------------------------------------------------
__global__ __launch_bounds__(256) void proto_final(
    const float* __restrict__ lossterm, const float* __restrict__ correct,
    float* __restrict__ out)
{
  __shared__ float sl[256];
  __shared__ float sc[256];
  float a = 0.f, c = 0.f;
  for (int i = threadIdx.x; i < N_ROWS; i += 256) {
    a += lossterm[i];
    c += correct[i];
  }
  sl[threadIdx.x] = a;
  sc[threadIdx.x] = c;
  __syncthreads();
  for (int s = 128; s > 0; s >>= 1) {
    if ((int)threadIdx.x < s) {
      sl[threadIdx.x] += sl[threadIdx.x + s];
      sc[threadIdx.x] += sc[threadIdx.x + s];
    }
    __syncthreads();
  }
  if (threadIdx.x == 0) {
    out[0] = sl[0] / (float)N_ROWS;
    out[1] = 100.0f * sc[0] / (float)N_ROWS;
  }
}

// ---------------------------------------------------------------------------
extern "C" void kernel_launch(void* const* d_in, const int* in_sizes, int n_in,
                              void* d_out, int out_size, void* d_ws, size_t ws_size,
                              hipStream_t stream)
{
  (void)in_sizes; (void)n_in; (void)out_size; (void)ws_size;
  const float* x = (const float*)d_in[0];

  char* base = (char*)d_ws;
  float* bvec     = (float*)(base);               //  8192 f32
  float* lossterm = (float*)(base + 32 * 1024);   //  8192 f32
  float* correct  = (float*)(base + 64 * 1024);   //  8192 f32
  const size_t elems = (size_t)N_ROWS * DFEAT;    // 1,572,864
  _Float16* ph = (_Float16*)(base + 128 * 1024);  // 3 MB each
  _Float16* pl = ph + elems;
  _Float16* ah = pl + elems;
  _Float16* al = ah + elems;

  proto_prep<<<N_ROWS / 8, 256, 0, stream>>>(x, ph, pl, ah, al, bvec);
  proto_gemm<<<(N_ROWS / 16) / 4, 128, 0, stream>>>(ph, pl, ah, al, bvec,
                                                    lossterm, correct);
  proto_final<<<1, 256, 0, stream>>>(lossterm, correct, (float*)d_out);
}
